// fBMGenerator_57621281243641
// MI455X (gfx1250) — matrix operand order repaired and checked by reading, not compiled, not run on hardware
//
#include <hip/hip_runtime.h>
#include <hip/hip_bf16.h>
#include <math.h>

typedef __attribute__((ext_vector_type(16))) __bf16 v16bf;
typedef __attribute__((ext_vector_type(8)))  float  v8f;

#define T_    1024
#define N_    1023   // T-1
#define PANEL 32

// One workgroup per batch. 1024 threads = 32 wave32 waves.
// Schur (Toeplitz Cholesky) recursion in double in LDS; every 32 columns,
// flush the L-panel through bf16x2 split-precision WMMA into per-wave
// f32 accumulators (dx tiles). Then scan (cumsum) in LDS and store.
__global__ __launch_bounds__(1024, 1) void fbm_schur_wmma_kernel(
    const float* __restrict__ alpha,
    const float* __restrict__ tau,
    const float* __restrict__ diffusion,
    const float* __restrict__ du,
    float* __restrict__ out)
{
    __shared__ double r_[T_];            // first Toeplitz column (lag k)
    __shared__ double u_[T_];            // Schur generator 1
    __shared__ double v_[T_];            // Schur generator 2
    __shared__ float  dus[T_][4];        // sqrt(D)-scaled noise, padded
    __shared__ __bf16 Phi[T_][PANEL];    // L panel, bf16 high part
    __shared__ __bf16 Plo[T_][PANEL];    // L panel, bf16 low residual
    __shared__ float  dxb[T_][4];        // dx rows (post-WMMA), then scanned

    const int b    = blockIdx.x;
    const int t    = threadIdx.x;        // row index 0..1023 (1023 = pad row)
    const int lane = t & 31;
    const int wave = t >> 5;

    const double a  = (double)alpha[b];
    const double tb = (double)tau[b];
    const float  sd = sqrtf(diffusion[b]);

    // ---- load & scale noise: du_s = du * sqrt(diffusion) ----
    if (t < N_) {
        const float* p = du + ((size_t)b * N_ + t) * 3;
        dus[t][0] = p[0] * sd;
        dus[t][1] = p[1] * sd;
        dus[t][2] = p[2] * sd;
        dus[t][3] = 0.0f;
    } else {
        dus[t][0] = dus[t][1] = dus[t][2] = dus[t][3] = 0.0f;
    }

    // ---- first column of covariance: r[k]=0.5((k+1)^a+|k-1|^a-2k^a), tapered ----
    {
        double R = 0.0;
        if (t < N_) {
            double k  = (double)t;
            double p1 = pow(k + 1.0, a);
            double p0 = (t == 0) ? 0.0 : pow(k, a);
            double pm = (t == 0) ? 1.0 : ((t == 1) ? 0.0 : pow(k - 1.0, a));
            R = 0.5 * (p1 + pm - 2.0 * p0);
            if (k >= tb) R *= exp(-(k - tb));   // LAM = 1
        }
        r_[t] = R;
    }
    __syncthreads();

    // ---- init generators; column 0 of L = r/sqrt(r0) -> panel slot 0 ----
    {
        double s0 = sqrt(r_[0]);
        double ui = r_[t] / s0;
        u_[t] = ui;
        v_[t] = (t == 0) ? 0.0 : ui;
        float  f = (float)ui;
        __bf16 h = (__bf16)f;
        Phi[t][0] = h;
        Plo[t][0] = (__bf16)(f - (float)h);
    }

    v8f acc0 = {0.f,0.f,0.f,0.f,0.f,0.f,0.f,0.f};   // strip = wave
    v8f acc1 = {0.f,0.f,0.f,0.f,0.f,0.f,0.f,0.f};   // strip = wave + 32
    __syncthreads();

    const int ncol  = lane & 15;                 // B/C/D: N = lane % 16
    const int kofsB = (lane < 16) ? 0  : 16;     // B: K half select
    const int khalf = (lane < 16) ? 0  : 8;      // A: K offset per lane half
    const int mrow  = lane & 15;                 // A: M = lane % 16

    // ---- Schur recursion over columns k = 1..n-1, fused panel WMMA flush ----
    for (int k = 1; k < N_; ++k) {
        // all LDS reads before the barrier
        double us = (t >= 1) ? u_[t - 1] : 0.0;  // down-shifted generator
        double vv = v_[t];
        double uk = u_[k - 1];                   // shifted u at position k
        double vk = v_[k];
        __syncthreads();

        double rho = vk / uk;                    // |rho| < 1 for SPD Toeplitz
        double cc  = 1.0 / sqrt(fmax(1.0 - rho * rho, 1e-300));
        double un  = (us - rho * vv) * cc;
        double vn  = (vv - rho * us) * cc;
        if (t < k) { un = 0.0; vn = 0.0; }       // lower-triangular column: zero above diagonal
        u_[t] = un;
        v_[t] = vn;

        const int slot = k & (PANEL - 1);
        {
            float  f = (float)un;
            __bf16 h = (__bf16)f;
            Phi[t][slot] = h;
            Plo[t][slot] = (__bf16)(f - (float)h);
        }

        const bool last = (k == N_ - 1);
        if (slot == PANEL - 1 || last) {
            if (last && slot != PANEL - 1) {     // zero unused tail of final panel
                for (int s = slot + 1; s < PANEL; ++s) {
                    Phi[t][s] = (__bf16)0.0f;
                    Plo[t][s] = (__bf16)0.0f;
                }
            }
            __syncthreads();
            const int kbase = k - slot;

            // B tile (32x16 bf16): b[e] -> (K = (e<8 ? e : e+8) + kofsB/2, N = ncol); dim-padded
            v16bf bhi, blo;
#pragma unroll
            for (int e = 0; e < 16; ++e) {
                float  x  = (ncol < 3) ? dus[kbase + ((e < 8) ? e : e + 8) + (kofsB >> 1)][ncol] : 0.0f;
                __bf16 xh = (__bf16)x;
                bhi[e] = xh;
                blo[e] = (__bf16)(x - (float)xh);
            }

            // two 16-row strips per wave
#pragma unroll
            for (int sidx = 0; sidx < 2; ++sidx) {
                const int row = (((wave + sidx * 32) << 4) + mrow);
                v16bf ahi, alo;
#pragma unroll
                for (int e = 0; e < 16; ++e) {
                    const int kk = ((e < 8) ? e : e + 8) + khalf;  // A 16x32 layout
                    ahi[e] = Phi[row][kk];
                    alo[e] = Plo[row][kk];
                }
                v8f acc = sidx ? acc1 : acc0;
                acc = __builtin_amdgcn_wmma_f32_16x16x32_bf16(
                        false, ahi, false, bhi, (short)0, acc, false, false);
                acc = __builtin_amdgcn_wmma_f32_16x16x32_bf16(
                        false, ahi, false, blo, (short)0, acc, false, false);
                acc = __builtin_amdgcn_wmma_f32_16x16x32_bf16(
                        false, alo, false, bhi, (short)0, acc, false, false);
                if (sidx) acc1 = acc; else acc0 = acc;
            }
        }
        __syncthreads();
    }

    // ---- spill C tiles to LDS: row = 16*strip + M, col = N (keep N<3) ----
#pragma unroll
    for (int sidx = 0; sidx < 2; ++sidx) {
        v8f acc = sidx ? acc1 : acc0;
        const int base = (((wave + sidx * 32) << 4)) + ((lane < 16) ? 0 : 8);
        if (ncol < 3) {
#pragma unroll
            for (int j = 0; j < 8; ++j)
                dxb[base + j][ncol] = acc[j];
        }
    }
    __syncthreads();

    // ---- inclusive scan (cumsum over time) of dx rows, in-place ----
    for (int off = 1; off < T_; off <<= 1) {
        float a0 = 0.f, a1 = 0.f, a2 = 0.f;
        if (t >= off) {
            a0 = dxb[t - off][0];
            a1 = dxb[t - off][1];
            a2 = dxb[t - off][2];
        }
        __syncthreads();
        if (t >= off) {
            dxb[t][0] += a0;
            dxb[t][1] += a1;
            dxb[t][2] += a2;
        }
        __syncthreads();
    }

    // ---- output: row 0 = zeros, row t+1 = cumsum through dx[t] ----
    float* o = out + (size_t)b * T_ * 3;
    if (t == 0) { o[0] = 0.f; o[1] = 0.f; o[2] = 0.f; }
    if (t < N_) {
        o[(t + 1) * 3 + 0] = dxb[t][0];
        o[(t + 1) * 3 + 1] = dxb[t][1];
        o[(t + 1) * 3 + 2] = dxb[t][2];
    }
}

extern "C" void kernel_launch(void* const* d_in, const int* in_sizes, int n_in,
                              void* d_out, int out_size, void* d_ws, size_t ws_size,
                              hipStream_t stream) {
    const float* alpha = (const float*)d_in[0];
    const float* tau   = (const float*)d_in[1];
    const float* diff  = (const float*)d_in[2];
    const float* du    = (const float*)d_in[3];
    // d_in[4] = T (scalar, fixed 1024 by setup)
    float* out = (float*)d_out;
    const int BS = in_sizes[0];

    fbm_schur_wmma_kernel<<<BS, 1024, 0, stream>>>(alpha, tau, diff, du, out);
}
